// STSGCN_71992241815935
// MI455X (gfx1250) — matrix operand order repaired and checked by reading, not compiled, not run on hardware
//
#include <hip/hip_runtime.h>
#include <math.h>

// ---------------------------------------------------------------------------
// STSGCN forward for MI455X (gfx1250). All GEMMs: v_wmma_f32_16x16x32_f16,
// 32x32 register-blocked per wave (4 WMMAs / K-chunk). All WMMA operands are
// stored in f16 (weights pre-transposed to NxK once per launch), so fragments
// load as two 16B v8h vectors each -> 8 global_load_b128 + 4 wmma per chunk,
// no cvt in the hot loop. f32 kept where numerics need it (scores, adj, o1).
// LSTM branch of the reference is dead w.r.t. d_out -> skipped.
// ---------------------------------------------------------------------------

typedef __attribute__((ext_vector_type(16))) _Float16 v16h;
typedef __attribute__((ext_vector_type(8)))  _Float16 v8h;
typedef __attribute__((ext_vector_type(8)))  float    v8f;

#define MP   608   // padded node dim (3*N=600 -> 608 = 19*32)
#define NN_  200   // N vertices
#define CC   64    // channels
#define ATT_ 128

constexpr int EPI_SCALE     = 0;  // C = acc * scale
constexpr int EPI_BIAS      = 1;  // C = acc + bias[n]
constexpr int EPI_BIAS_RELU = 2;  // C = relu(acc + bias[n])
constexpr int EPI_SIGMOID   = 3;  // C = sigmoid(acc * scale)

__device__ __forceinline__ v16h cat8(v8h lo, v8h hi) {
    return __builtin_shufflevector(lo, hi, 0, 1, 2, 3, 4, 5, 6, 7,
                                   8, 9, 10, 11, 12, 13, 14, 15);
}

// One wave computes a 32x32 tile of C = A * B^T; K consumed in chunks of 32.
// A: M x K row-major f16 (lda). B: N x K row-major f16 (ldb)  [always TB].
// C: f16 or f32 per OUTF16 (ldc); batch strides in elements of each buffer.
template<int EPI, bool OUTF16>
__global__ __launch_bounds__(32)
void wmma_gemm32(const _Float16* __restrict__ A, const _Float16* __restrict__ B,
                 const float* __restrict__ bias, void* __restrict__ Cv,
                 int K, int lda, int ldb, int ldc,
                 long long sA, long long sB, long long sC, float scale)
{
    A += (long long)blockIdx.z * sA;
    B += (long long)blockIdx.z * sB;
    const int lane = threadIdx.x;
    const int half = lane >> 4;       // K-half selector (ISA 7.12.2)
    const int r    = lane & 15;       // A row / B col within 16-tile
    const int tm   = blockIdx.y * 32;
    const int tn   = blockIdx.x * 32;
    const _Float16* Ar0 = A + (long long)(tm + r) * lda;        // M-tile 0 row
    const _Float16* Ar1 = Ar0 + 16LL * lda;                     // M-tile 1 row
    const _Float16* Br0 = B + (long long)(tn + r) * ldb;        // N-tile 0 col
    const _Float16* Br1 = Br0 + 16LL * ldb;                     // N-tile 1 col

    v8f acc00 = {}, acc01 = {}, acc10 = {}, acc11 = {};
    for (int kc = 0; kc < K; kc += 32) {
        if (kc + 32 < K) {            // hide L2 latency of the next chunk
            __builtin_prefetch(Ar0 + kc + 32, 0, 1);
            __builtin_prefetch(Ar1 + kc + 32, 0, 1);
        }
        // A frag: a[0..7] = K(kc+8h)+0..7 ; a[8..15] = K(kc+16+8h)+0..7
        const int kaA = kc + 8 * half;
        v16h a0 = cat8(*(const v8h*)(Ar0 + kaA), *(const v8h*)(Ar0 + kaA + 16));
        v16h a1 = cat8(*(const v8h*)(Ar1 + kaA), *(const v8h*)(Ar1 + kaA + 16));
        // B frag: b[0..7] = K(kc+8h)+0..7 ; b[8..15] = K(kc+16+8h)+0..7
        const int kaB = kc + 8 * half;
        v16h b0 = cat8(*(const v8h*)(Br0 + kaB), *(const v8h*)(Br0 + kaB + 16));
        v16h b1 = cat8(*(const v8h*)(Br1 + kaB), *(const v8h*)(Br1 + kaB + 16));

        acc00 = __builtin_amdgcn_wmma_f32_16x16x32_f16(false, a0, false, b0,
                                                       (short)0, acc00, false, false);
        acc01 = __builtin_amdgcn_wmma_f32_16x16x32_f16(false, a0, false, b1,
                                                       (short)0, acc01, false, false);
        acc10 = __builtin_amdgcn_wmma_f32_16x16x32_f16(false, a1, false, b0,
                                                       (short)0, acc10, false, false);
        acc11 = __builtin_amdgcn_wmma_f32_16x16x32_f16(false, a1, false, b1,
                                                       (short)0, acc11, false, false);
    }

    _Float16* C16 = (_Float16*)Cv + (long long)blockIdx.z * sC;
    float*    C32 = (float*)Cv    + (long long)blockIdx.z * sC;
    auto store_tile = [&](const v8f& acc, int m0, int n0) {
#pragma unroll
        for (int i = 0; i < 8; ++i) {
            const int m = m0 + i + 8 * half;   // C/D layout: VGPR i -> M = i+8*half
            const int n = n0 + r;
            float v = acc[i];
            if (EPI == EPI_SCALE)          v = v * scale;
            else if (EPI == EPI_BIAS)      v = v + bias[n];
            else if (EPI == EPI_BIAS_RELU) v = fmaxf(v + bias[n], 0.f);
            else /* EPI_SIGMOID */         v = 1.f / (1.f + __expf(-v * scale));
            if (OUTF16) C16[(long long)m * ldc + n] = (_Float16)v;
            else        C32[(long long)m * ldc + n] = v;
        }
    };
    store_tile(acc00, tm,      tn);
    store_tile(acc01, tm,      tn + 16);
    store_tile(acc10, tm + 16, tn);
    store_tile(acc11, tm + 16, tn + 16);
}

// dst (N x K, f16) = transpose of src (K x N, f32)
__global__ void transpose_w(const float* __restrict__ src, _Float16* __restrict__ dst,
                            int K, int N, int total)
{
    int idx = blockIdx.x * blockDim.x + threadIdx.x;
    if (idx >= total) return;
    const int k = idx % K, n = idx / K;
    dst[idx] = (_Float16)src[(long long)k * N + n];
}

// vT[bw][n][m] = relu(qkv[bw][m][256+n]) ; n in [0,128), m in [0,608)
__global__ void vtrans_relu(const _Float16* __restrict__ qkv, _Float16* __restrict__ vt,
                            int total)
{
    int idx = blockIdx.x * blockDim.x + threadIdx.x;
    if (idx >= total) return;
    const int m  = idx % MP;
    const int n  = (idx / MP) % ATT_;
    const int bw = idx / (MP * ATT_);
    float f = (float)qkv[((long long)bw * MP + m) * 384 + 256 + n];
    vt[idx] = (_Float16)fmaxf(f, 0.f);
}

// Windowed input: O[(b*nw+w)][j*200+n][c] = X[b][w+j][n][c]+temb[w+j][c]+semb[n][c]
// pad rows 600..607 zeroed.  (TIN = float for layer 1, _Float16 for layer 2)
template<typename TIN>
__global__ void window_embed(const TIN* __restrict__ X, const float* __restrict__ temb,
                             const float* __restrict__ semb, _Float16* __restrict__ O,
                             int T, int nw, int total)
{
    int idx = blockIdx.x * blockDim.x + threadIdx.x;
    if (idx >= total) return;
    const int c  = idx & 63;
    const int m  = (idx >> 6) % MP;
    const int bw = idx / (MP * CC);
    const int w  = bw % nw, b = bw / nw;
    float val = 0.f;
    if (m < 600) {
        const int j = m / NN_, n = m % NN_;
        const int t = w + j;
        val = (float)X[(((long long)b * T + t) * NN_ + n) * CC + c]
            + temb[t * CC + c] + semb[n * CC + c];
    }
    O[idx] = (_Float16)val;
}

// node[m][c] = mean over BW batches of cur[bw][m][c]  (f16 in/out, f32 accum)
__global__ void node_mean(const _Float16* __restrict__ cur, _Float16* __restrict__ node,
                          int BW)
{
    int idx = blockIdx.x * blockDim.x + threadIdx.x;
    if (idx >= MP * CC) return;
    float s = 0.f;
    for (int i = 0; i < BW; ++i) s += (float)cur[(long long)i * MP * CC + idx];
    node[idx] = (_Float16)(s / (float)BW);
}

// Per row: softmax over cols[0,600) * adj, renormalized; writes f16 w; pad cols 0.
// w[c] = e[c]*adj[c] / (sum(e*adj) + 1e-8*sum(e))   [== ref aw/(ssum+1e-8)*mask]
__global__ __launch_bounds__(256)
void softmax_adj(const float* __restrict__ sc, const float* __restrict__ adj,
                 _Float16* __restrict__ wout)
{
    const int row = blockIdx.x % MP;
    const float* s = sc + (long long)blockIdx.x * MP;
    _Float16* wrow = wout + (long long)blockIdx.x * MP;
    const float* arow = adj + (long long)row * MP;
    __shared__ float r1[256], r2[256];
    const int tid = threadIdx.x;

    float m = -3.4e38f;
    for (int c = tid; c < 600; c += 256) m = fmaxf(m, s[c]);
    r1[tid] = m; __syncthreads();
    for (int st = 128; st > 0; st >>= 1) {
        if (tid < st) r1[tid] = fmaxf(r1[tid], r1[tid + st]);
        __syncthreads();
    }
    m = r1[0]; __syncthreads();

    float se = 0.f, sa = 0.f;
    for (int c = tid; c < 600; c += 256) {
        const float e = __expf(s[c] - m);
        se += e; sa += e * arow[c];
    }
    r1[tid] = se; r2[tid] = sa; __syncthreads();
    for (int st = 128; st > 0; st >>= 1) {
        if (tid < st) { r1[tid] += r1[tid + st]; r2[tid] += r2[tid + st]; }
        __syncthreads();
    }
    se = r1[0]; sa = r2[0]; __syncthreads();
    const float inv = (sa > 0.f) ? 1.f / (sa + 1e-8f * se) : 0.f;

    for (int c = tid; c < MP; c += 256) {
        float w = 0.f;
        if (c < 600) w = __expf(s[c] - m) * arow[c] * inv;
        wrow[c] = (_Float16)w;
    }
}

// Running max over filters of cur rows [400,600): cm[bw][n][c]  (f16)
__global__ void cand_max(const _Float16* __restrict__ cur, _Float16* __restrict__ cm,
                         int total, int first)
{
    int idx = blockIdx.x * blockDim.x + threadIdx.x;
    if (idx >= total) return;
    const int c  = idx & 63;
    const int n  = (idx >> 6) % NN_;
    const int bw = idx / (NN_ * CC);
    const _Float16 v = cur[((long long)bw * MP + 400 + n) * CC + c];
    if (first) cm[idx] = v;
    else       cm[idx] = (cm[idx] > v) ? cm[idx] : v;
}

// d[(b*200+n)][t*64+c] = h2[b][t][n][c]   (800 x 512, f16)
__global__ void build_d(const _Float16* __restrict__ h2, _Float16* __restrict__ d, int total)
{
    int idx = blockIdx.x * blockDim.x + threadIdx.x;
    if (idx >= total) return;
    const int col = idx & 511, row = idx >> 9;
    const int t = col >> 6, c = col & 63;
    const int b = row / NN_, n = row % NN_;
    d[idx] = h2[(((long long)b * 8 + t) * NN_ + n) * CC + c];
}

// out[b][p*200+n][f] = o1[b*200+n][f], p in [0,12)
__global__ void tile_out(const float* __restrict__ o1, float* __restrict__ out, int total)
{
    int idx = blockIdx.x * blockDim.x + threadIdx.x;
    if (idx >= total) return;
    const int f   = idx & 127;
    const int row = (idx >> 7) % 2400;
    const int b   = idx / (2400 * 128);
    const int n   = row % NN_;
    out[idx] = o1[((long long)b * NN_ + n) * 128 + f];
}

extern "C" void kernel_launch(void* const* d_in, const int* in_sizes, int n_in,
                              void* d_out, int out_size, void* d_ws, size_t ws_size,
                              hipStream_t stream)
{
    (void)in_sizes; (void)n_in; (void)out_size; (void)ws_size;
    const float* x     = (const float*)d_in[0];
    const float* Wqkv  = (const float*)d_in[4];
    const float* bqkv  = (const float*)d_in[5];
    const float* Wlin  = (const float*)d_in[6];
    const float* blin  = (const float*)d_in[7];
    const float* Wa    = (const float*)d_in[8];
    const float* Wb    = (const float*)d_in[9];
    const float* temb0 = (const float*)d_in[13];
    const float* temb1 = (const float*)d_in[14];
    const float* semb  = (const float*)d_in[15];
    const float* Wo    = (const float*)d_in[16];
    const float* bo    = (const float*)d_in[17];

    // ---- workspace carve-out (256B-aligned slices; BW max = 40) ----
    char* base = (char*)d_ws;
    size_t off = 0;
    auto alloc = [&](size_t bytes) -> void* {
        void* p = base + off;
        off += (bytes + 255) & ~(size_t)255;
        return p;
    };
    _Float16* curh   = (_Float16*)alloc(40LL * MP * CC * 2);
    _Float16* nxth   = (_Float16*)alloc(40LL * MP * CC * 2);
    _Float16* qkvh   = (_Float16*)alloc(40LL * MP * 384 * 2);   // [q|k|v]
    _Float16* vT     = (_Float16*)alloc(40LL * ATT_ * MP * 2);  // relu(v)^T
    float*    scores = (float*)   alloc(40LL * MP * MP * 4);
    _Float16* wbuf   = (_Float16*)alloc(40LL * MP * MP * 2);
    _Float16* att    = (_Float16*)alloc(40LL * MP * ATT_ * 2);
    float*    adjb   = (float*)   alloc((size_t)MP * MP * 4);
    _Float16* node   = (_Float16*)alloc((size_t)MP * CC * 2);
    _Float16* Pb     = (_Float16*)alloc((size_t)MP * CC * 2);
    _Float16* Qb     = (_Float16*)alloc((size_t)MP * CC * 2);
    _Float16* cmax1  = (_Float16*)alloc(40LL * NN_ * CC * 2);   // (B,10,200,64)
    _Float16* cmax2  = (_Float16*)alloc(32LL * NN_ * CC * 2);   // (B, 8,200,64)
    _Float16* dbuf   = (_Float16*)alloc(800LL * 512 * 2);
    float*    o1     = (float*)   alloc(800LL * 128 * 4);
    _Float16* WqkvT  = (_Float16*)alloc(384LL * 64 * 2);
    _Float16* WlinT  = (_Float16*)alloc(64LL * 128 * 2);
    _Float16* WaT    = (_Float16*)alloc(64LL * 64 * 2);
    _Float16* WbT    = (_Float16*)alloc(64LL * 64 * 2);
    _Float16* WoT    = (_Float16*)alloc(128LL * 512 * 2);

    auto cdiv = [](int a, int b) { return (a + b - 1) / b; };
    const float inv_sqrt_att = 0.088388347648318440550f; // 1/sqrt(128)

    // One-time weight transposes to f16 NxK
    transpose_w<<<cdiv(64 * 384, 256), 256, 0, stream>>>(Wqkv, WqkvT, 64, 384, 64 * 384);
    transpose_w<<<cdiv(128 * 64, 256), 256, 0, stream>>>(Wlin, WlinT, 128, 64, 128 * 64);
    transpose_w<<<cdiv(64 * 64, 256), 256, 0, stream>>>(Wa, WaT, 64, 64, 64 * 64);
    transpose_w<<<cdiv(64 * 64, 256), 256, 0, stream>>>(Wb, WbT, 64, 64, 64 * 64);
    transpose_w<<<cdiv(512 * 128, 256), 256, 0, stream>>>(Wo, WoT, 512, 128, 512 * 128);

    _Float16* cur = curh;
    _Float16* nxt = nxth;

    auto run_filter_loop = [&](int BW) {
        const long long sX = (long long)MP * CC;
        const long long sQ = (long long)MP * 384;
        const long long sS = (long long)MP * MP;
        const long long sV = (long long)MP * ATT_;
        const long long sT = (long long)ATT_ * MP;
        for (int f = 0; f < 3; ++f) {
            node_mean<<<cdiv(MP * CC, 256), 256, 0, stream>>>(cur, node, BW);

            // P = node @ Wa, Q = node @ Wb    (608 x 64, K=64)
            wmma_gemm32<EPI_SCALE, true><<<dim3(2, 19, 1), 32, 0, stream>>>(
                node, WaT, nullptr, Pb, 64, 64, 64, 64, 0, 0, 0, 1.f);
            wmma_gemm32<EPI_SCALE, true><<<dim3(2, 19, 1), 32, 0, stream>>>(
                node, WbT, nullptr, Qb, 64, 64, 64, 64, 0, 0, 0, 1.f);
            // adj = sigmoid(P Q^T / 8)        (608 x 608, K=64) -> f32
            wmma_gemm32<EPI_SIGMOID, false><<<dim3(19, 19, 1), 32, 0, stream>>>(
                Pb, Qb, nullptr, adjb, 64, 64, 64, MP, 0, 0, 0, 0.125f);

            // qkv = cur @ Wqkv + bqkv         (batched 608 x 384, K=64) -> f16
            wmma_gemm32<EPI_BIAS, true><<<dim3(12, 19, BW), 32, 0, stream>>>(
                cur, WqkvT, bqkv, qkvh, 64, 64, 64, 384, sX, 0, sQ, 1.f);

            // scores = q k^T / sqrt(128)      (batched 608 x 608, K=128) -> f32
            wmma_gemm32<EPI_SCALE, false><<<dim3(19, 19, BW), 32, 0, stream>>>(
                qkvh, qkvh + 128, nullptr, scores, 128, 384, 384, MP, sQ, sQ, sS,
                inv_sqrt_att);

            // w = renorm(softmax(scores) * adj) -> f16
            softmax_adj<<<BW * MP, 256, 0, stream>>>(scores, adjb, wbuf);

            // vT = relu(v)^T                  (batched 128 x 608)
            const int vtot = BW * ATT_ * MP;
            vtrans_relu<<<cdiv(vtot, 256), 256, 0, stream>>>(qkvh, vT, vtot);

            // att = w @ relu(v)               (batched 608 x 128, K=608) -> f16
            wmma_gemm32<EPI_SCALE, true><<<dim3(4, 19, BW), 32, 0, stream>>>(
                wbuf, vT, nullptr, att, MP, MP, MP, ATT_, sS, sT, sV, 1.f);

            // nxt = att @ Wlin + blin         (batched 608 x 64, K=128) -> f16
            wmma_gemm32<EPI_BIAS, true><<<dim3(2, 19, BW), 32, 0, stream>>>(
                att, WlinT, blin, nxt, 128, 128, 128, 64, sV, 0, sX, 1.f);

            _Float16* cmax = (BW == 40) ? cmax1 : cmax2;
            const int ctot = BW * NN_ * CC;
            cand_max<<<cdiv(ctot, 256), 256, 0, stream>>>(nxt, cmax, ctot, f == 0 ? 1 : 0);

            _Float16* t = cur; cur = nxt; nxt = t;
        }
    };

    // Layer 1: (B,12,200,64) -> (B,10,200,64)
    {
        const int T = 12, nw = 10, BW = 40;
        const int wtot = BW * MP * CC;
        window_embed<float><<<cdiv(wtot, 256), 256, 0, stream>>>(
            x, temb0, semb, cur, T, nw, wtot);
        run_filter_loop(BW);
    }
    // Layer 2: (B,10,200,64) -> (B,8,200,64)
    {
        const int T = 10, nw = 8, BW = 32;
        const int wtot = BW * MP * CC;
        window_embed<_Float16><<<cdiv(wtot, 256), 256, 0, stream>>>(
            cmax1, temb1, semb, cur, T, nw, wtot);
        run_filter_loop(BW);
    }

    build_d<<<cdiv(800 * 512, 256), 256, 0, stream>>>(cmax2, dbuf, 800 * 512);
    // o1 = relu(d @ Wo + bo)   (800 x 128, K=512) -> f32
    wmma_gemm32<EPI_BIAS_RELU, false><<<dim3(4, 25, 1), 32, 0, stream>>>(
        dbuf, WoT, bo, o1, 512, 512, 512, 128, 0, 0, 0, 1.f);
    tile_out<<<cdiv(4 * 2400 * 128, 256), 256, 0, stream>>>(o1, (float*)d_out, 4 * 2400 * 128);
}
